// DotAttention_44332652429980
// MI455X (gfx1250) — matrix operand order repaired and checked by reading, not compiled, not run on hardware
//
#include <hip/hip_runtime.h>
#include <hip/hip_bf16.h>

// Problem constants (from reference)
#define B_    8
#define JX_   2048
#define JM_   2048
#define DIN_  150
#define H_    96
#define G_    300
#define RPAD_ 320   // res row padded to 320 (10 x K=32 bf16 WMMA steps)
#define DPAD_ 160   // D_MEM padded to 160 (10 x N=16 tiles)

typedef __attribute__((ext_vector_type(16))) __bf16 v16bf;
typedef __attribute__((ext_vector_type(8)))  float  v8f;
typedef __attribute__((ext_vector_type(4)))  unsigned uint32x4;
typedef __attribute__((ext_vector_type(8)))  int      int32x8;
typedef __attribute__((ext_vector_type(4)))  int      int32x4;

union BF16x16 {
  v16bf v;
  __bf16 e[16];
  uint4 q[2];
};

static __device__ __forceinline__ __bf16 f2bf(float f) {
  union { float f; unsigned u; } a; a.f = f;
  unsigned r = a.u + 0x7FFFu + ((a.u >> 16) & 1u);  // round-to-nearest-even
  union { unsigned short s; __bf16 b; } o; o.s = (unsigned short)(r >> 16);
  return o.b;
}

// A-fragment (16x32 bf16): lane m holds K = kb..kb+7 and kb+16..kb+23 where
// kb = 8*(lane>=16). Caller passes p already offset to (row, kchunk + kb).
static __device__ __forceinline__ v16bf load_a_frag(const __bf16* p) {
  BF16x16 u;
  u.q[0] = *reinterpret_cast<const uint4*>(p);
  u.q[1] = *reinterpret_cast<const uint4*>(p + 16);
  return u.v;
}
// B-fragment (32x16 bf16): lane n holds K = kb..kb+7 and kb+16..kb+23 (kb = 8*(lane>=16)).
static __device__ __forceinline__ v16bf load_b_frag(const __bf16* p) {
  BF16x16 u;
  u.q[0] = *reinterpret_cast<const uint4*>(p);
  u.q[1] = *reinterpret_cast<const uint4*>(p + 16);
  return u.v;
}

// ---- Tensor Data Mover: 2D bf16 tile -> LDS, LDS-padded 4 DW per 16 DW ----
// pad_enable=1, pad_interval=3 (16 DWORDs), pad_amount=3 (4 DWORDs):
// every 64B of payload is followed by 16B of LDS padding (bank-conflict-free
// ds_load_b128 fragment reads afterwards).
static __device__ __forceinline__ void tdm_load_tile_2d(
    unsigned lds_addr, const void* gptr, unsigned tile_d0, unsigned tile_d1,
    unsigned long long stride0 /* elements */) {
  const unsigned long long ga = (unsigned long long)(size_t)gptr;
  uint32x4 g0;
  g0[0] = 1u;                                  // count=1 (valid user descriptor)
  g0[1] = lds_addr;                            // LDS byte address
  g0[2] = (unsigned)ga;                        // global_addr[31:0]
  g0[3] = (unsigned)((ga >> 32) & 0x01FFFFFFu) | 0x80000000u;  // addr[56:32]|type=2
  unsigned d[8];
  d[0] = (1u << 16) | (1u << 20) | (3u << 22) | (3u << 25);
  //      data=2B     pad_en       16DW intvl    4DW pad ; workgroup_mask=0
  d[1] = 0u;                    // atomic barrier off; tensor_dim0[15:0]=0
  d[2] = 0x10u;                 // tensor_dim0 = 1<<20 (hi), tensor_dim1[15:0]=0
  d[3] = 0x10u | (tile_d0 << 16);  // tensor_dim1 = 1<<20 (hi), tile_dim0
  d[4] = tile_d1 & 0xFFFFu;        // tile_dim1 ; tile_dim2 = 0
  d[5] = (unsigned)stride0;        // tensor_dim0_stride[31:0]
  d[6] = (unsigned)((stride0 >> 32) & 0xFFFFu);  // stride[47:32]; dim1_stride=0
  d[7] = 0u;
  int32x8 g1;
#pragma unroll
  for (int i = 0; i < 8; ++i) g1[i] = (int)d[i];
  int32x4 z4;
  z4[0] = 0; z4[1] = 0; z4[2] = 0; z4[3] = 0;
  int32x8 z8;
#pragma unroll
  for (int i = 0; i < 8; ++i) z8[i] = 0;
  // 6-arg toolchain variant: (g0, g1, g2, g3, extra, cpol)
  __builtin_amdgcn_tensor_load_to_lds(g0, g1, z4, z4, z8, 0);
}

static __device__ __forceinline__ unsigned lds_byte_off(const void* p) {
  return (unsigned)(size_t)p;  // low 32 bits of flat LDS address = LDS address
}

// ---------------- Kernel 1: project inputs, seed res ----------------
__global__ __launch_bounds__(192) void proj_inputs_kernel(
    const float* __restrict__ inputs, const float* __restrict__ Wi,
    const float* __restrict__ bi, __bf16* __restrict__ inp_b,
    float* __restrict__ res) {
  const int b = blockIdx.x >> 11;
  const int x = blockIdx.x & 2047;
  const int t = threadIdx.x;
  const float* row = inputs + (size_t)(b * JX_ + x) * DIN_;
  if (t < H_) {
    float acc = bi[t];
#pragma unroll 5
    for (int d = 0; d < DIN_; ++d) acc = __fmaf_rn(row[d], Wi[d * H_ + t], acc);
    acc = fmaxf(acc, 0.f);
    inp_b[(size_t)(b * JX_ + x) * H_ + t] = f2bf(acc);
  }
  float* rrow = res + (size_t)(b * JX_ + x) * RPAD_;
  if (t < DIN_) rrow[t] = row[t];                       // res[:,0:150] = inputs
  if (t >= DIN_ && t < DIN_ + (RPAD_ - G_)) rrow[G_ + (t - DIN_)] = 0.f;  // pad
}

// -------- Kernel 2: project memory, build Vt (transposed, padded), mask -----
__global__ __launch_bounds__(160) void proj_memory_kernel(
    const float* __restrict__ memory, const float* __restrict__ Wm,
    const float* __restrict__ bm, const int* __restrict__ mask,
    __bf16* __restrict__ mem_b, __bf16* __restrict__ Vt,
    float* __restrict__ maskf) {
  const int b = blockIdx.x >> 11;
  const int y = blockIdx.x & 2047;
  const int t = threadIdx.x;
  const float* row = memory + (size_t)(b * JM_ + y) * DIN_;
  if (t < H_) {
    float acc = bm[t];
#pragma unroll 5
    for (int d = 0; d < DIN_; ++d) acc = __fmaf_rn(row[d], Wm[d * H_ + t], acc);
    acc = fmaxf(acc, 0.f);
    mem_b[(size_t)(b * JM_ + y) * H_ + t] = f2bf(acc);
  }
  if (t < DIN_)  Vt[((size_t)b * DPAD_ + t) * JM_ + y] = f2bf(row[t]);
  if (t >= DIN_ && t < DPAD_) Vt[((size_t)b * DPAD_ + t) * JM_ + y] = f2bf(0.f);
  if (t == 0) maskf[b * JM_ + y] = mask[b * JM_ + y] ? 0.f : -1e30f;
}

// ---------------- Kernel 3: Wg -> transposed padded bf16 ----------------
__global__ __launch_bounds__(256) void prep_wg_kernel(
    const float* __restrict__ Wg, __bf16* __restrict__ WgT) {
  const int idx = blockIdx.x * 256 + threadIdx.x;
  if (idx >= RPAD_ * RPAD_) return;
  const int n = idx / RPAD_, k = idx % RPAD_;
  const float v = (n < G_ && k < G_) ? Wg[(size_t)k * G_ + n] : 0.f;
  WgT[idx] = f2bf(v);
}

// ---------------- Kernel 4: flash attention via WMMA (S^T form) -------------
// 1 wave = 16 JX columns; 8 waves/block share (b, jm-chunk) A-operands.
// TDM double-buffers memory_/Vt chunks into LDS (one issuing wave,
// s_wait_tensorcnt + block barrier); softmax axis is register-resident.
__global__ __launch_bounds__(256) void attn_kernel(
    const __bf16* __restrict__ inp_b, const __bf16* __restrict__ mem_b,
    const __bf16* __restrict__ Vt, const float* __restrict__ maskf,
    float* __restrict__ res) {
  // Padded LDS tiles: mem rows 96 elems -> 120-elem (240B) stride,
  // Vt rows 32 elems -> 40-elem (80B) stride (TDM pad: 4 DW per 16 DW).
  __shared__ __attribute__((aligned(32))) __bf16 mem_s[2][32 * 120 + 8];
  __shared__ __attribute__((aligned(32))) __bf16 vt_s[2][160 * 40 + 8];

  const int lane = threadIdx.x & 31;
  const int wave = threadIdx.x >> 5;
  const int tile = blockIdx.x * 8 + wave;   // global 16-col jx tile
  const int b   = tile >> 7;                // 128 tiles per batch
  const int jx0 = (tile & 127) << 4;
  const int nl = lane & 15;
  const int hi = lane >> 4;

  const __bf16* inpb = inp_b + (size_t)(b * JX_ + jx0) * H_;
  const __bf16* memb = mem_b + (size_t)b * JM_ * H_;
  const __bf16* Vtb  = Vt + (size_t)b * DPAD_ * JM_;
  const float*  mfb  = maskf + b * JM_;

  // B-fragments of inputs_^T (fixed per wave): B(k,n)=inputs_[jx0+n][k]
  v16bf binp[3];
#pragma unroll
  for (int f = 0; f < 3; ++f)
    binp[f] = load_b_frag(inpb + (size_t)nl * H_ + f * 32 + hi * 8);

  v8f o[10];
#pragma unroll
  for (int t = 0; t < 10; ++t)
#pragma unroll
    for (int r = 0; r < 8; ++r) o[t][r] = 0.f;

  float m_run = -INFINITY, l_run = 0.f;
  const float scale = 0.10206207262f;  // 1/sqrt(96)

  // Prologue: wave 0 issues TDM for chunk 0 into buffer 0.
  if (wave == 0) {
    tdm_load_tile_2d(lds_byte_off(&mem_s[0][0]), memb, 96u, 32u, 96ull);
    tdm_load_tile_2d(lds_byte_off(&vt_s[0][0]), Vtb, 32u, 160u, (unsigned long long)JM_);
  }

  for (int ci = 0; ci < JM_ / 32; ++ci) {
    const int jm0 = ci * 32;
    const int pb = ci & 1;
    if (wave == 0) __builtin_amdgcn_s_wait_tensorcnt(0);  // chunk ci landed
    __syncthreads();  // all waves done with buffer pb^1; chunk ci visible
    if (wave == 0 && ci + 1 < JM_ / 32) {  // prefetch chunk ci+1 into pb^1
      const int jn = jm0 + 32;
      tdm_load_tile_2d(lds_byte_off(&mem_s[pb ^ 1][0]),
                       memb + (size_t)jn * H_, 96u, 32u, 96ull);
      tdm_load_tile_2d(lds_byte_off(&vt_s[pb ^ 1][0]),
                       Vtb + jn, 32u, 160u, (unsigned long long)JM_);
    }

    // Two 16x16 S^T tiles from LDS: rows jm0..+15 / jm0+16..+31, cols jx tile
    v8f c0, c1;
#pragma unroll
    for (int r = 0; r < 8; ++r) { c0[r] = 0.f; c1[r] = 0.f; }
#pragma unroll
    for (int f = 0; f < 3; ++f) {
      v16bf a0 = load_a_frag(&mem_s[pb][nl * 120 + f * 40 + hi * 8]);
      c0 = __builtin_amdgcn_wmma_f32_16x16x32_bf16(false, a0, false, binp[f],
                                                   (short)0, c0, false, false);
    }
#pragma unroll
    for (int f = 0; f < 3; ++f) {
      v16bf a1 = load_a_frag(&mem_s[pb][(16 + nl) * 120 + f * 40 + hi * 8]);
      c1 = __builtin_amdgcn_wmma_f32_16x16x32_bf16(false, a1, false, binp[f],
                                                   (short)0, c1, false, false);
    }
    // scale + additive mask (mask indexed by jm = M row of S^T)
    float s0[8], s1[8];
#pragma unroll
    for (int r = 0; r < 8; ++r) {
      s0[r] = c0[r] * scale + mfb[jm0 + hi * 8 + r];
      s1[r] = c1[r] * scale + mfb[jm0 + 16 + hi * 8 + r];
    }
    // online softmax: per-lane max/sum over 16 values + partner-lane combine
    float mx = s0[0];
#pragma unroll
    for (int r = 1; r < 8; ++r) mx = fmaxf(mx, s0[r]);
#pragma unroll
    for (int r = 0; r < 8; ++r) mx = fmaxf(mx, s1[r]);
    mx = fmaxf(mx, __shfl_xor(mx, 16, 32));
    const float m_new = fmaxf(m_run, mx);
    const float alpha = __expf(m_run - m_new);
    float p0[8], p1[8], sum = 0.f;
#pragma unroll
    for (int r = 0; r < 8; ++r) {
      p0[r] = __expf(s0[r] - m_new);
      p1[r] = __expf(s1[r] - m_new);
      sum += p0[r] + p1[r];
    }
    sum += __shfl_xor(sum, 16, 32);
    l_run = l_run * alpha + sum;
    m_run = m_new;
#pragma unroll
    for (int t = 0; t < 10; ++t)
#pragma unroll
      for (int r = 0; r < 8; ++r) o[t][r] *= alpha;

    // Build P^T B-fragment (K=32) from this lane's own p0/p1; the xor-16 q0/q1 go unused.
    float q0[8], q1[8];
#pragma unroll
    for (int r = 0; r < 8; ++r) {
      q0[r] = __shfl_xor(p0[r], 16, 32);
      q1[r] = __shfl_xor(p1[r], 16, 32);
    }
    BF16x16 pbf;
#pragma unroll
    for (int i = 0; i < 8; ++i) {
      pbf.e[i]     = f2bf(p0[i]);
      pbf.e[8 + i] = f2bf(p1[i]);
    }
    // O^T += V^T . P^T  (10 N-tiles of d, K=32), V^T tiles from LDS
#pragma unroll
    for (int t = 0; t < 10; ++t) {
      v16bf av = load_a_frag(&vt_s[pb][(t * 16 + nl) * 40 + hi * 8]);
      o[t] = __builtin_amdgcn_wmma_f32_16x16x32_bf16(false, av, false, pbf.v,
                                                     (short)0, o[t], false, false);
    }
  }

  const float rcp = 1.f / l_run;
  float* rrow = res + (size_t)(b * JX_ + jx0 + nl) * RPAD_ + DIN_;
#pragma unroll
  for (int t = 0; t < 10; ++t)
#pragma unroll
    for (int r = 0; r < 8; ++r) {
      const int d = t * 16 + hi * 8 + r;
      if (d < DIN_) rrow[d] = o[t][r] * rcp;
    }
}

// ---------------- Kernel 5: gating GEMM + sigmoid + multiply ----------------
__global__ __launch_bounds__(256) void gate_kernel(
    const float* __restrict__ res, const __bf16* __restrict__ WgT,
    const float* __restrict__ bg, float* __restrict__ out) {
  const int lane = threadIdx.x & 31;
  const int wave = threadIdx.x >> 5;
  const int tile = blockIdx.x * 8 + wave;
  const int b   = tile >> 7;
  const int jx0 = (tile & 127) << 4;
  const int nl = lane & 15;
  const int hi = lane >> 4;

  // A-fragments: 16 rows of res (f32 -> bf16), K padded to 320
  const float* rowp = res + (size_t)(b * JX_ + jx0 + nl) * RPAD_;
  v16bf af[10];
#pragma unroll
  for (int kf = 0; kf < 10; ++kf) {
    const int base = kf * 32 + hi * 8;
    const float4 x0 = *reinterpret_cast<const float4*>(rowp + base);
    const float4 x1 = *reinterpret_cast<const float4*>(rowp + base + 4);
    const float4 y0 = *reinterpret_cast<const float4*>(rowp + base + 16);
    const float4 y1 = *reinterpret_cast<const float4*>(rowp + base + 20);
    BF16x16 u;
    u.e[0]  = f2bf(x0.x); u.e[1]  = f2bf(x0.y); u.e[2]  = f2bf(x0.z); u.e[3]  = f2bf(x0.w);
    u.e[4]  = f2bf(x1.x); u.e[5]  = f2bf(x1.y); u.e[6]  = f2bf(x1.z); u.e[7]  = f2bf(x1.w);
    u.e[8]  = f2bf(y0.x); u.e[9]  = f2bf(y0.y); u.e[10] = f2bf(y0.z); u.e[11] = f2bf(y0.w);
    u.e[12] = f2bf(y1.x); u.e[13] = f2bf(y1.y); u.e[14] = f2bf(y1.z); u.e[15] = f2bf(y1.w);
    af[kf] = u.v;
  }

  for (int nt = 0; nt < 19; ++nt) {   // 19*16 = 304 >= 300 output cols
    v8f acc;
#pragma unroll
    for (int r = 0; r < 8; ++r) acc[r] = 0.f;
    const __bf16* wrow = WgT + (size_t)(nt * 16 + nl) * RPAD_ + hi * 8;
#pragma unroll
    for (int kf = 0; kf < 10; ++kf) {
      v16bf bw = load_b_frag(wrow + kf * 32);
      acc = __builtin_amdgcn_wmma_f32_16x16x32_bf16(false, af[kf], false, bw,
                                                    (short)0, acc, false, false);
    }
    const int n = nt * 16 + nl;
    if (n < G_) {
      const float bgv = bg[n];
#pragma unroll
      for (int r = 0; r < 8; ++r) {
        const int m = r + 8 * hi;
        const size_t rowi = (size_t)(b * JX_ + jx0 + m);
        const float rv = res[rowi * RPAD_ + n];
        const float g = 1.f / (1.f + __expf(-(acc[r] + bgv)));
        out[rowi * G_ + n] = g * rv;
      }
    }
  }
}

// ---------------------------- launcher ----------------------------
extern "C" void kernel_launch(void* const* d_in, const int* in_sizes, int n_in,
                              void* d_out, int out_size, void* d_ws, size_t ws_size,
                              hipStream_t stream) {
  (void)in_sizes; (void)n_in; (void)out_size; (void)ws_size;
  const float* inputs = (const float*)d_in[0];
  const float* memory = (const float*)d_in[1];
  const int*   mask   = (const int*)d_in[2];
  const float* Wi     = (const float*)d_in[3];
  const float* bi     = (const float*)d_in[4];
  const float* Wm     = (const float*)d_in[5];
  const float* bm     = (const float*)d_in[6];
  const float* Wg     = (const float*)d_in[7];
  const float* bg     = (const float*)d_in[8];
  float* out = (float*)d_out;

  char* ws = (char*)d_ws;
  __bf16* inp_b = (__bf16*)(ws + 0);          //  3,145,728 B  [B,JX,96] bf16
  __bf16* mem_b = (__bf16*)(ws + 3145728);    //  3,145,728 B  [B,JM,96] bf16
  __bf16* Vt    = (__bf16*)(ws + 6291456);    //  5,242,880 B  [B,160,JM] bf16
  float*  maskf = (float*)(ws + 11534336);    //     65,536 B  [B,JM] f32
  float*  res   = (float*)(ws + 11599872);    // 20,971,520 B  [B,JX,320] f32
  __bf16* WgT   = (__bf16*)(ws + 32571392);   //    204,800 B  [320,320] bf16

  proj_inputs_kernel<<<B_ * JX_, 192, 0, stream>>>(inputs, Wi, bi, inp_b, res);
  proj_memory_kernel<<<B_ * JM_, 160, 0, stream>>>(memory, Wm, bm, mask,
                                                   mem_b, Vt, maskf);
  prep_wg_kernel<<<(RPAD_ * RPAD_ + 255) / 256, 256, 0, stream>>>(Wg, WgT);
  attn_kernel<<<(B_ * JX_) / (16 * 8), 256, 0, stream>>>(inp_b, mem_b, Vt,
                                                         maskf, res);
  gate_kernel<<<(B_ * JX_) / (16 * 8), 256, 0, stream>>>(res, WgT, bg, out);
}
